// EdgeWeightGNN_42210938585394
// MI455X (gfx1250) — matrix operand order repaired and checked by reading, not compiled, not run on hardware
//
#include <hip/hip_runtime.h>
#include <hip/hip_bf16.h>

// EdgeWeightGNN for MI455X (gfx1250, wave32).
// Heavy edge matmuls (msg: M x 64 x 64 per step, decoder L1: M x 133 x 64) run on
// v_wmma_f32_16x16x32_f16: one wave owns a 16-edge tile, K=64 via chained WMMAs.
// B fragments live in a per-block LDS table and are re-read per jt iteration
// (jt loop is NOT unrolled so the DS loads stay loop-variant -> LICM cannot
// hoist them into long-lived registers -> no scratch spills in the hot loop).

typedef __attribute__((ext_vector_type(16))) _Float16 v16h;
typedef __attribute__((ext_vector_type(8)))  _Float16 v8h;
typedef __attribute__((ext_vector_type(8)))  float    v8f;

#define H 64
#define ND 12
#define ED 5
#define GD 11
#define EWN 32

// ---------------------------------------------------------------------------
// WMMA fragment loaders (CDNA5 wave32 layouts, CDNA5 ISA 7.12.2)
// A (16x32 f16): lanes 0-15 row M=lane hold K = kc*32 + {0..7, 16..23};
//                lanes 16-31 row M=lane-16 hold K = kc*32 + {8..15, 24..31}.
// ---------------------------------------------------------------------------
__device__ __forceinline__ v16h load_a_frag(const _Float16* row, int kc, bool hi) {
  int kb = kc * 32 + (hi ? 8 : 0);
  v8h lo = *(const v8h*)(row + kb);        // 16B global load
  v8h hg = *(const v8h*)(row + kb + 16);   // 16B global load
  v16h a;
#pragma unroll
  for (int i = 0; i < 8; ++i) { a[i] = lo[i]; a[8 + i] = hg[i]; }
  return a;
}

__device__ __forceinline__ v8f wmma_f16(v16h a, v16h b, v8f c) {
  return __builtin_amdgcn_wmma_f32_16x16x32_f16(false, a, false, b, (short)0, c,
                                                false, false);
}

// Fill a per-lane B-fragment LDS table from Wt (64x64 f16, element (k,n) at k*64+n).
// B (32x16 f16): lanes 0-15 column N=lane hold K = kc*32 + {0..7, 16..23};
//                lanes 16-31 column N=lane-16 hold K = kc*32 + {8..15, 24..31}.
// Table layout: [frag][lane][16 halfs] (32B contiguous per (frag,lane)).
__device__ __forceinline__ void fill_b_table(_Float16 (*tbl)[32][16],
                                             const _Float16* Wt, int fragBase,
                                             int nFrags, int tid, int nthreads) {
  for (int pairIdx = tid; pairIdx < nFrags * 32; pairIdx += nthreads) {
    int f = pairIdx >> 5, ln = pairIdx & 31;
    int jt = f >> 1, kc = f & 1;
    int n  = jt * 16 + (ln & 15);
    int kb = kc * 32 + ((ln & 16) ? 8 : 0);
#pragma unroll
    for (int i = 0; i < 16; ++i) tbl[fragBase + f][ln][i] = Wt[(kb + (i < 8 ? i : i + 8)) * H + n];
  }
}

// ---------------------------------------------------------------------------
// Weight preparation: f16-transpose msg_W and dec_W1 halves; f32-transpose GRU mats.
// ---------------------------------------------------------------------------
__global__ void prep_weights_kernel(const float* __restrict__ msg_W,
                                    const float* __restrict__ dec_W1,
                                    const float* __restrict__ Wih,
                                    const float* __restrict__ Whh,
                                    _Float16* msgWt, _Float16* dW1s, _Float16* dW1d,
                                    float* WihT, float* WhhT) {
  int idx = blockIdx.x * blockDim.x + threadIdx.x;
  if (idx < H * H) {
    int k = idx >> 6, j = idx & 63;
    msgWt[idx] = (_Float16)msg_W[j * H + k];
    dW1s[idx]  = (_Float16)dec_W1[j * (2 * H + ED) + k];
    dW1d[idx]  = (_Float16)dec_W1[j * (2 * H + ED) + H + k];
  }
  if (idx < 3 * H * H) {                 // 12288: WihT[k*192 + row] = Wih[row*64 + k]
    int k = idx / 192, r = idx % 192;
    WihT[idx] = Wih[r * H + k];
    WhhT[idx] = Whh[r * H + k];
  }
}

// ---------------------------------------------------------------------------
// h = tanh([x, u] @ W_in.T + b_in); also write f16 mirror.
// ---------------------------------------------------------------------------
__global__ void inproj_kernel(const float* __restrict__ x, const float* __restrict__ u,
                              const float* __restrict__ W, const float* __restrict__ b,
                              float* hA, _Float16* h16, int n) {
  int idx = blockIdx.x * blockDim.x + threadIdx.x;
  int node = idx >> 6, j = idx & 63;
  if (node >= n) return;
  float s = b[j];
  const float* wr = W + j * (ND + GD);
#pragma unroll
  for (int q = 0; q < ND; ++q) s += x[node * ND + q] * wr[q];
#pragma unroll
  for (int q = 0; q < GD; ++q) s += u[q] * wr[ND + q];
  float v = tanhf(s);
  hA[idx]  = v;
  h16[idx] = (_Float16)v;
}

// ---------------------------------------------------------------------------
// alpha = sigmoid(relu([ea,u] @ W1.T + b1) @ W2.T + b2). u-part folded once per block.
// ---------------------------------------------------------------------------
__global__ void alpha_kernel(const float* __restrict__ edge_attr, const float* __restrict__ u,
                             const float* __restrict__ W1, const float* __restrict__ b1,
                             const float* __restrict__ W2, const float* __restrict__ b2,
                             float* alpha, long m) {
  __shared__ float sW1e[EWN][ED];
  __shared__ float sHu[EWN];
  __shared__ float sW2[EWN];
  __shared__ float sb2;
  int t = threadIdx.x;
  if (t < EWN) {
    float s = b1[t];
#pragma unroll
    for (int q = 0; q < GD; ++q) s += u[q] * W1[t * (ED + GD) + ED + q];
    sHu[t] = s;
    sW2[t] = W2[t];
#pragma unroll
    for (int q = 0; q < ED; ++q) sW1e[t][q] = W1[t * (ED + GD) + q];
  }
  if (t == 0) sb2 = b2[0];
  __syncthreads();
  long e = (long)blockIdx.x * blockDim.x + t;
  if (e >= m) return;
  float ef[ED];
#pragma unroll
  for (int q = 0; q < ED; ++q) ef[q] = edge_attr[e * ED + q];
  float acc = sb2;
#pragma unroll 4
  for (int i = 0; i < EWN; ++i) {
    float s = sHu[i];
#pragma unroll
    for (int q = 0; q < ED; ++q) s += ef[q] * sW1e[i][q];
    acc += fmaxf(s, 0.f) * sW2[i];
  }
  alpha[e] = 1.f / (1.f + expf(-acc));
}

__global__ void degree_kernel(const int* __restrict__ dstI, float* deg, long m) {
  long e = (long)blockIdx.x * blockDim.x + threadIdx.x;
  if (e < m) atomicAdd(deg + dstI[e], 1.f);
}

// ---------------------------------------------------------------------------
// msg step: per 16-edge tile, D = h16[src] @ msgWt (WMMA, bias seeded in C), then
// agg[dst] += alpha * relu(D) via f32 atomics (agg is L2-resident, 12.8 MB).
// ---------------------------------------------------------------------------
__global__ void msg_kernel(const _Float16* __restrict__ h16, const _Float16* __restrict__ Wt,
                           const float* __restrict__ alpha,
                           const int* __restrict__ srcI, const int* __restrict__ dstI,
                           const float* __restrict__ msg_b,
                           float* agg, int numTiles) {
  __shared__ __align__(32) _Float16 sFragB[8][32][16];  // 8 KB
  __shared__ float sMb[H];
  fill_b_table(sFragB, Wt, 0, 8, threadIdx.x, blockDim.x);
  for (int i = threadIdx.x; i < H; i += blockDim.x) sMb[i] = msg_b[i];
  __syncthreads();

  int lane = threadIdx.x & 31;
  bool hi  = (lane & 16) != 0;
  int wave = (blockIdx.x * blockDim.x + threadIdx.x) >> 5;
  int nw   = (gridDim.x * blockDim.x) >> 5;

#pragma unroll 1
  for (int t = wave; t < numTiles; t += nw) {
    long e_self = (long)t * 16 + (lane & 15);
    int  sv = srcI[e_self];
    int  dv = dstI[e_self];
    float av_self = alpha[e_self];
    const _Float16* row = h16 + (long)sv * H;
    v16h a0 = load_a_frag(row, 0, hi);
    v16h a1 = load_a_frag(row, 1, hi);

    // Per-row scalars, invariant across jt: hoist the cross-lane broadcasts.
    float avr[8];
    int   ddr[8];
#pragma unroll
    for (int r = 0; r < 8; ++r) {
      int m  = r + (hi ? 8 : 0);
      avr[r] = __shfl(av_self, m, 32);
      ddr[r] = __shfl(dv, m, 32);
    }

    // jt NOT unrolled: DS fragment loads stay loop-variant (no LICM hoist,
    // no spills); only 2 B fragments live at a time.
#pragma unroll 1
    for (int jt = 0; jt < 4; ++jt) {
      int j = jt * 16 + (lane & 15);
      float mbv = sMb[j];
      v8f c;
#pragma unroll
      for (int r = 0; r < 8; ++r) c[r] = mbv;     // seed C with bias
      v16h b0 = *(const v16h*)(&sFragB[jt * 2 + 0][lane][0]);
      v16h b1 = *(const v16h*)(&sFragB[jt * 2 + 1][lane][0]);
      c = wmma_f16(a0, b0, c);
      c = wmma_f16(a1, b1, c);
#pragma unroll
      for (int r = 0; r < 8; ++r) {
        float v = fmaxf(c[r], 0.f);
        unsigned off = (unsigned)ddr[r] * H + j;  // N*H < 2^31: 32-bit offset
        atomicAdd(&agg[off], avr[r] * v);
      }
    }
  }
}

// ---------------------------------------------------------------------------
// GRU: per (node, j) thread, rows staged in LDS, transposed weights (coalesced).
// ---------------------------------------------------------------------------
__global__ void gru_kernel(const float* __restrict__ agg, const float* __restrict__ deg,
                           const float* __restrict__ h_in,
                           const float* __restrict__ WihT, const float* __restrict__ bih,
                           const float* __restrict__ WhhT, const float* __restrict__ bhh,
                           float* h_out, _Float16* h16, int n) {
  __shared__ float sa[4][H];
  __shared__ float sh[4][H];
  int j    = threadIdx.x & 63;
  int slot = threadIdx.x >> 6;
  int node = blockIdx.x * 4 + slot;
  if (node < n) {
    float dg = fmaxf(deg[node], 1.f);
    sa[slot][j] = agg[(long)node * H + j] / dg;
    sh[slot][j] = h_in[(long)node * H + j];
  }
  __syncthreads();
  if (node >= n) return;
  float gir = bih[j], giz = bih[H + j], gin = bih[2 * H + j];
  float ghr = bhh[j], ghz = bhh[H + j], ghn = bhh[2 * H + j];
#pragma unroll 4
  for (int k = 0; k < H; ++k) {
    float av = sa[slot][k], hv = sh[slot][k];
    const float* wi = WihT + k * 192;
    const float* wh = WhhT + k * 192;
    gir += av * wi[j];       ghr += hv * wh[j];
    giz += av * wi[H + j];   ghz += hv * wh[H + j];
    gin += av * wi[2*H + j]; ghn += hv * wh[2*H + j];
  }
  float r  = 1.f / (1.f + expf(-(gir + ghr)));
  float z  = 1.f / (1.f + expf(-(giz + ghz)));
  float ng = tanhf(gin + r * ghn);
  float hn = (1.f - z) * ng + z * sh[slot][j];
  h_out[(long)node * H + j] = hn;
  h16[(long)node * H + j]   = (_Float16)hn;
}

// ---------------------------------------------------------------------------
// Decoder: L1 via WMMA (src K=64 + dst K=64 + edge_attr/bias seed in C).
// B fragments in a 16 KB per-block LDS table, re-read per jt iteration;
// relu tile -> padded per-wave LDS slice; L2/L3 on VALU; float4 store to (M,4).
// ---------------------------------------------------------------------------
__global__ void dec_kernel(const _Float16* __restrict__ h16,
                           const _Float16* __restrict__ B1sG, const _Float16* __restrict__ B1dG,
                           const int* __restrict__ srcI, const int* __restrict__ dstI,
                           const float* __restrict__ edge_attr,
                           const float* __restrict__ dec_W1, const float* __restrict__ dec_b1,
                           const float* __restrict__ dec_W2, const float* __restrict__ dec_b2,
                           const float* __restrict__ dec_W3, const float* __restrict__ dec_b3,
                           float* __restrict__ out, int numTiles) {
  // frag index f = mat*8 + jt*2 + kc; per-lane 16 halfs (32B) contiguous.
  __shared__ __align__(32) _Float16 sFragB[16][32][16];  // 16 KB
  __shared__ float sW1e[H][ED];
  __shared__ float sb1[H];
  __shared__ float sW2[32][65];   // padded: bank-conflict-free
  __shared__ float sb2[32];
  __shared__ float sW3[4][32];
  __shared__ float sb3[4];
  __shared__ float y1[8][16][65]; // per-wave relu(L1) tile, padded

  int t0 = threadIdx.x;
  fill_b_table(sFragB, B1sG, 0, 8, t0, blockDim.x);
  fill_b_table(sFragB, B1dG, 8, 8, t0, blockDim.x);
  for (int i = t0; i < H * ED; i += blockDim.x)
    sW1e[i / ED][i % ED] = dec_W1[(i / ED) * (2 * H + ED) + 2 * H + (i % ED)];
  for (int i = t0; i < H;      i += blockDim.x) sb1[i] = dec_b1[i];
  for (int i = t0; i < 32 * H; i += blockDim.x) sW2[i >> 6][i & 63] = dec_W2[i];
  for (int i = t0; i < 32;     i += blockDim.x) sb2[i] = dec_b2[i];
  for (int i = t0; i < 128;    i += blockDim.x) sW3[i >> 5][i & 31] = dec_W3[i];
  for (int i = t0; i < 4;      i += blockDim.x) sb3[i] = dec_b3[i];
  __syncthreads();

  int lane  = t0 & 31;
  bool hi   = (lane & 16) != 0;
  int wslot = t0 >> 5;
  int wave  = (blockIdx.x * blockDim.x + t0) >> 5;
  int nw    = (gridDim.x * blockDim.x) >> 5;

#pragma unroll 1
  for (int t = wave; t < numTiles; t += nw) {
    long e_self = (long)t * 16 + (lane & 15);
    int  sv = srcI[e_self], dv = dstI[e_self];
    const _Float16* rs = h16 + (long)sv * H;
    const _Float16* rd = h16 + (long)dv * H;
    v16h as0 = load_a_frag(rs, 0, hi), as1 = load_a_frag(rs, 1, hi);
    v16h ad0 = load_a_frag(rd, 0, hi), ad1 = load_a_frag(rd, 1, hi);

    // Edge-attr broadcasts, invariant across jt: hoist out of the jt loop.
    float ea[ED];
#pragma unroll
    for (int q = 0; q < ED; ++q) ea[q] = edge_attr[e_self * ED + q];
    float eam[8][ED];
#pragma unroll
    for (int r = 0; r < 8; ++r) {
      int m = r + (hi ? 8 : 0);
#pragma unroll
      for (int q = 0; q < ED; ++q) eam[r][q] = __shfl(ea[q], m, 32);
    }

    // jt NOT unrolled: keeps only 4 B fragments live, DS loads per iteration.
#pragma unroll 1
    for (int jt = 0; jt < 4; ++jt) {
      int j = jt * 16 + (lane & 15);
      float w1q[ED];
#pragma unroll
      for (int q = 0; q < ED; ++q) w1q[q] = sW1e[j][q];
      float b1v = sb1[j];
      v8f c;
#pragma unroll
      for (int r = 0; r < 8; ++r) {          // seed C with bias + edge_attr part
        float s = b1v;
#pragma unroll
        for (int q = 0; q < ED; ++q) s += eam[r][q] * w1q[q];
        c[r] = s;
      }
      v16h bs0 = *(const v16h*)(&sFragB[jt * 2 + 0][lane][0]);
      v16h bs1 = *(const v16h*)(&sFragB[jt * 2 + 1][lane][0]);
      v16h bd0 = *(const v16h*)(&sFragB[8 + jt * 2 + 0][lane][0]);
      v16h bd1 = *(const v16h*)(&sFragB[8 + jt * 2 + 1][lane][0]);
      c = wmma_f16(as0, bs0, c);
      c = wmma_f16(as1, bs1, c);
      c = wmma_f16(ad0, bd0, c);
      c = wmma_f16(ad1, bd1, c);
#pragma unroll
      for (int r = 0; r < 8; ++r) {
        int m = r + (hi ? 8 : 0);
        y1[wslot][m][j] = fmaxf(c[r], 0.f);  // same-wave LDS: DS in-order, no barrier
      }
    }

    // Layers 2+3: 2 lanes per edge (lane, lane^16) split the 32 hidden units.
    int m   = lane & 15;
    int j2b = hi ? 16 : 0;
    float acc2[16];
#pragma unroll
    for (int jj = 0; jj < 16; ++jj) acc2[jj] = sb2[j2b + jj];
#pragma unroll 2
    for (int k = 0; k < H; ++k) {
      float yv = y1[wslot][m][k];
#pragma unroll
      for (int jj = 0; jj < 16; ++jj) acc2[jj] += yv * sW2[j2b + jj][k];
    }
#pragma unroll
    for (int jj = 0; jj < 16; ++jj) acc2[jj] = fmaxf(acc2[jj], 0.f);
    float o[4];
#pragma unroll
    for (int tt = 0; tt < 4; ++tt) {
      float p = 0.f;
#pragma unroll
      for (int jj = 0; jj < 16; ++jj) p += acc2[jj] * sW3[tt][j2b + jj];
      p += __shfl_xor(p, 16, 32);
      o[tt] = p + sb3[tt];
    }
    if (!hi) {
      float4 ov = make_float4(o[0], o[1], o[2], o[3]);
      *(float4*)(out + e_self * 4) = ov;
    }
  }
}

// ---------------------------------------------------------------------------
extern "C" void kernel_launch(void* const* d_in, const int* in_sizes, int n_in,
                              void* d_out, int out_size, void* d_ws, size_t ws_size,
                              hipStream_t stream) {
  const float* x        = (const float*)d_in[0];
  const int*   ei       = (const int*)d_in[1];    // (2, M) int
  const float* edge_attr= (const float*)d_in[2];
  const float* u        = (const float*)d_in[3];
  const float* W_in     = (const float*)d_in[4];
  const float* b_in     = (const float*)d_in[5];
  const float* ewn_W1   = (const float*)d_in[6];
  const float* ewn_b1   = (const float*)d_in[7];
  const float* ewn_W2   = (const float*)d_in[8];
  const float* ewn_b2   = (const float*)d_in[9];
  const float* msg_W    = (const float*)d_in[10];
  const float* msg_b    = (const float*)d_in[11];
  const float* gru_Wih  = (const float*)d_in[12];
  const float* gru_bih  = (const float*)d_in[13];
  const float* gru_Whh  = (const float*)d_in[14];
  const float* gru_bhh  = (const float*)d_in[15];
  const float* dec_W1   = (const float*)d_in[16];
  const float* dec_b1   = (const float*)d_in[17];
  const float* dec_W2   = (const float*)d_in[18];
  const float* dec_b2   = (const float*)d_in[19];
  const float* dec_W3   = (const float*)d_in[20];
  const float* dec_b3   = (const float*)d_in[21];

  const int  N = in_sizes[0] / ND;
  const long M = in_sizes[2] / ED;
  const int* srcI = ei;
  const int* dstI = ei + M;

  char* p = (char*)d_ws;
  auto carve = [&p](size_t bytes) -> void* {
    void* r = (void*)p;
    p += (bytes + 255) & ~(size_t)255;
    return r;
  };
  float*     hA    = (float*)carve((size_t)N * H * 4);
  float*     hB    = (float*)carve((size_t)N * H * 4);
  _Float16*  h16   = (_Float16*)carve((size_t)N * H * 2);
  float*     agg   = (float*)carve((size_t)N * H * 4);
  float*     deg   = (float*)carve((size_t)N * 4);
  float*     alpha = (float*)carve((size_t)M * 4);
  _Float16*  msgWt = (_Float16*)carve((size_t)H * H * 2);
  _Float16*  dW1s  = (_Float16*)carve((size_t)H * H * 2);
  _Float16*  dW1d  = (_Float16*)carve((size_t)H * H * 2);
  float*     WihT  = (float*)carve((size_t)3 * H * H * 4);
  float*     WhhT  = (float*)carve((size_t)3 * H * H * 4);

  hipMemsetAsync(deg, 0, (size_t)N * 4, stream);
  prep_weights_kernel<<<48, 256, 0, stream>>>(msg_W, dec_W1, gru_Wih, gru_Whh,
                                              msgWt, dW1s, dW1d, WihT, WhhT);
  inproj_kernel<<<(N * H + 255) / 256, 256, 0, stream>>>(x, u, W_in, b_in, hA, h16, N);
  alpha_kernel<<<(int)((M + 255) / 256), 256, 0, stream>>>(edge_attr, u, ewn_W1, ewn_b1,
                                                           ewn_W2, ewn_b2, alpha, M);
  degree_kernel<<<(int)((M + 255) / 256), 256, 0, stream>>>(dstI, deg, M);

  const int numTiles = (int)(M / 16);  // M = 1,250,000 -> 78125 exact tiles
  float* hcur = hA;
  float* hnext = hB;
  for (int s = 0; s < 3; ++s) {
    hipMemsetAsync(agg, 0, (size_t)N * H * 4, stream);
    msg_kernel<<<1024, 256, 0, stream>>>(h16, msgWt, alpha, srcI, dstI, msg_b, agg, numTiles);
    gru_kernel<<<(N + 3) / 4, 256, 0, stream>>>(agg, deg, hcur, WihT, gru_bih,
                                                WhhT, gru_bhh, hnext, h16, N);
    float* tmp = hcur; hcur = hnext; hnext = tmp;
  }

  dec_kernel<<<1024, 256, 0, stream>>>(h16, dW1s, dW1d, srcI, dstI, edge_attr,
                                       dec_W1, dec_b1, dec_W2, dec_b2, dec_W3, dec_b3,
                                       (float*)d_out, numTiles);
}
